// SinkhornKnopp_im_75411035783855
// MI455X (gfx1250) — compile-verified, matrix operand order checked by reading, not run on hardware
//
#include <hip/hip_runtime.h>
#include <hip/hip_bf16.h>

typedef __attribute__((ext_vector_type(16))) __bf16          v16bf;
typedef __attribute__((ext_vector_type(8)))  float           v8f;
typedef __attribute__((ext_vector_type(8)))  unsigned short  us8;
typedef __attribute__((ext_vector_type(16))) unsigned short  us16;

#define B_ROWS   32768
#define D_DIM    768
#define K_COLS   1000
#define K_PAD    1024
#define EPS_INV  20.0f

// ---------- bf16 helpers (round-to-nearest-even, bit-exact, no __bf16 arithmetic) ----------
__device__ __forceinline__ unsigned short f2bf(float f) {
    unsigned u = __builtin_bit_cast(unsigned, f);
    u += 0x7fffu + ((u >> 16) & 1u);
    return (unsigned short)(u >> 16);
}
__device__ __forceinline__ float bf2f(unsigned short s) {
    unsigned u = ((unsigned)s) << 16;
    return __builtin_bit_cast(float, u);
}

// ---------- prep: features row-L2-normalize, split into bf16 hi/lo ----------
__global__ __launch_bounds__(256)
void prep_feat_kernel(const float* __restrict__ F,
                      unsigned short* __restrict__ Ahi,
                      unsigned short* __restrict__ Alo) {
    __shared__ float red[256];
    const int b   = blockIdx.x;
    const int tid = threadIdx.x;
    const float* row = F + (size_t)b * D_DIM;
    float ss = 0.0f;
    for (int d = tid; d < D_DIM; d += 256) { float v = row[d]; ss += v * v; }
    red[tid] = ss; __syncthreads();
    for (int s = 128; s > 0; s >>= 1) { if (tid < s) red[tid] += red[tid + s]; __syncthreads(); }
    const float scale = 1.0f / fmaxf(sqrtf(red[0]), 1e-12f);
    for (int d = tid; d < D_DIM; d += 256) {
        float v = row[d] * scale;
        unsigned short hi = f2bf(v);
        unsigned short lo = f2bf(v - bf2f(hi));
        Ahi[(size_t)b * D_DIM + d] = hi;
        Alo[(size_t)b * D_DIM + d] = lo;
    }
}

// ---------- prep: head row-L2-normalize (axis=1, length 1000), transpose to [K_PAD][768], split ----------
__global__ __launch_bounds__(256)
void prep_head_kernel(const float* __restrict__ H,
                      unsigned short* __restrict__ Bhi,
                      unsigned short* __restrict__ Blo) {
    __shared__ float red[256];
    const int d   = blockIdx.x;              // 0..767
    const int tid = threadIdx.x;
    const float* row = H + (size_t)d * K_COLS;
    float ss = 0.0f;
    for (int n = tid; n < K_COLS; n += 256) { float v = row[n]; ss += v * v; }
    red[tid] = ss; __syncthreads();
    for (int s = 128; s > 0; s >>= 1) { if (tid < s) red[tid] += red[tid + s]; __syncthreads(); }
    const float scale = 1.0f / fmaxf(sqrtf(red[0]), 1e-12f);
    for (int n = tid; n < K_PAD; n += 256) {
        float v = (n < K_COLS) ? row[n] * scale : 0.0f;
        unsigned short hi = f2bf(v);
        unsigned short lo = f2bf(v - bf2f(hi));
        Bhi[(size_t)n * D_DIM + d] = hi;
        Blo[(size_t)n * D_DIM + d] = lo;
    }
}

// ---------- softmax over w (1x1000) ----------
__global__ __launch_bounds__(1024)
void softmax_w_kernel(const float* __restrict__ w, float* __restrict__ K2) {
    __shared__ float red[1024];
    const int tid = threadIdx.x;
    float x = (tid < K_COLS) ? w[tid] : -3.4e38f;
    red[tid] = x; __syncthreads();
    for (int s = 512; s > 0; s >>= 1) { if (tid < s) red[tid] = fmaxf(red[tid], red[tid + s]); __syncthreads(); }
    const float mx = red[0]; __syncthreads();
    float e = (tid < K_COLS) ? expf(x - mx) : 0.0f;
    red[tid] = e; __syncthreads();
    for (int s = 512; s > 0; s >>= 1) { if (tid < s) red[tid] += red[tid + s]; __syncthreads(); }
    if (tid < K_COLS) K2[tid] = e / red[0];
}

// ---------- GEMM + exp: Q0[b,k] = exp(20 * (A_norm @ Hs_norm)[b,k]) ----------
// bf16 hi/lo split: A*B ~= Ah*Bh + Ah*Bl + Al*Bh  (fp32-quality product via v_wmma_f32_16x16x32_bf16)
__device__ __forceinline__ v16bf load_ab16(const unsigned short* __restrict__ base) {
    us8 c0 = *(const us8*)(base);        // K = h*8 .. h*8+7
    us8 c1 = *(const us8*)(base + 16);   // K = 16+h*8 .. 16+h*8+7
    us16 t = __builtin_shufflevector(c0, c1, 0,1,2,3,4,5,6,7,8,9,10,11,12,13,14,15);
    return __builtin_bit_cast(v16bf, t);
}

__global__ __launch_bounds__(256)
void gemm_exp_kernel(const unsigned short* __restrict__ Ahi,
                     const unsigned short* __restrict__ Alo,
                     const unsigned short* __restrict__ Bhi,
                     const unsigned short* __restrict__ Blo,
                     float* __restrict__ Q0) {
    const int lane = threadIdx.x & 31;
    const int wave = threadIdx.x >> 5;
    const int r = lane & 15;     // row-in-tile for A, col-in-tile for B
    const int h = lane >> 4;     // K-half selector
    const int m0 = blockIdx.x * 32;
    const int n0 = (blockIdx.y * 8 + wave) * 32;

    v8f acc[2][2] = {};

    for (int kk = 0; kk < D_DIM; kk += 32) {
        v16bf ah[2], al[2], bh[2], bl[2];
        #pragma unroll
        for (int mt = 0; mt < 2; ++mt) {
            size_t base = (size_t)(m0 + mt * 16 + r) * D_DIM + kk + h * 8;
            ah[mt] = load_ab16(Ahi + base);
            al[mt] = load_ab16(Alo + base);
        }
        #pragma unroll
        for (int nt = 0; nt < 2; ++nt) {
            size_t base = (size_t)(n0 + nt * 16 + r) * D_DIM + kk + h * 8;
            bh[nt] = load_ab16(Bhi + base);
            bl[nt] = load_ab16(Blo + base);
        }
        // three passes keep 4 independent WMMAs between accumulator reuses
        #pragma unroll
        for (int mt = 0; mt < 2; ++mt)
            #pragma unroll
            for (int nt = 0; nt < 2; ++nt)
                acc[mt][nt] = __builtin_amdgcn_wmma_f32_16x16x32_bf16(
                    false, ah[mt], false, bh[nt], (short)0, acc[mt][nt], false, false);
        #pragma unroll
        for (int mt = 0; mt < 2; ++mt)
            #pragma unroll
            for (int nt = 0; nt < 2; ++nt)
                acc[mt][nt] = __builtin_amdgcn_wmma_f32_16x16x32_bf16(
                    false, ah[mt], false, bl[nt], (short)0, acc[mt][nt], false, false);
        #pragma unroll
        for (int mt = 0; mt < 2; ++mt)
            #pragma unroll
            for (int nt = 0; nt < 2; ++nt)
                acc[mt][nt] = __builtin_amdgcn_wmma_f32_16x16x32_bf16(
                    false, al[mt], false, bh[nt], (short)0, acc[mt][nt], false, false);
    }

    // Epilogue: C/D layout — VGPR j: lanes0-15 M=j, lanes16-31 M=8+j; N = lane&15
    #pragma unroll
    for (int mt = 0; mt < 2; ++mt) {
        #pragma unroll
        for (int nt = 0; nt < 2; ++nt) {
            const int n = n0 + nt * 16 + r;
            if (n < K_COLS) {
                const int mbase = m0 + mt * 16 + h * 8;
                #pragma unroll
                for (int j = 0; j < 8; ++j)
                    Q0[(size_t)(mbase + j) * K_COLS + n] = __expf(acc[mt][nt][j] * EPS_INV);
            }
        }
    }
}

// ---------- Sinkhorn reductions ----------
__global__ __launch_bounds__(1024)
void zero_c_kernel(float* __restrict__ C) { C[threadIdx.x] = 0.0f; }

// C[k] += sum over 256 rows of Q0[b,k] * rb[b]   (grid.x = 128 blocks of 256 rows)
__global__ __launch_bounds__(256)
void colsum_kernel(const float* __restrict__ Q0, const float* __restrict__ rb,
                   float* __restrict__ C, int useRb) {
    const int k  = threadIdx.x;           // 0..255
    const int b0 = blockIdx.x * 256;
    float a0 = 0.f, a1 = 0.f, a2 = 0.f, a3 = 0.f;
    for (int b = b0; b < b0 + 256; ++b) {
        const float r = useRb ? rb[b] : 1.0f;
        const float* row = Q0 + (size_t)b * K_COLS;
        a0 += row[k] * r;
        a1 += row[k + 256] * r;
        a2 += row[k + 512] * r;
        if (k < K_COLS - 768) a3 += row[k + 768] * r;
    }
    atomicAdd(&C[k], a0);
    atomicAdd(&C[k + 256], a1);
    atomicAdd(&C[k + 512], a2);
    if (k < K_COLS - 768) atomicAdd(&C[k + 768], a3);
}

__global__ __launch_bounds__(1024)
void scale_ck_kernel(const float* __restrict__ K2, const float* __restrict__ C,
                     float* __restrict__ ck) {
    const int k = threadIdx.x;
    if (k < K_COLS) ck[k] = K2[k] / C[k];
}

// one wave per row: rb[b] = 1 / (B * sum_k Q0[b,k]*ck[k])
__global__ __launch_bounds__(256)
void rowsum_kernel(const float* __restrict__ Q0, const float* __restrict__ ck,
                   float* __restrict__ rb) {
    const int lane = threadIdx.x & 31;
    const int b = blockIdx.x * 8 + (threadIdx.x >> 5);
    const float* row = Q0 + (size_t)b * K_COLS;
    float acc = 0.0f;
    for (int k = lane; k < K_COLS; k += 32) acc += row[k] * ck[k];
    for (int off = 16; off > 0; off >>= 1) acc += __shfl_xor(acc, off, 32);
    if (lane == 0) rb[b] = 1.0f / (32768.0f * acc);
}

// out[b,k] = Q0[b,k] * ck3[k] / R3[b]
__global__ __launch_bounds__(256)
void finalize_kernel(const float* __restrict__ Q0, const float* __restrict__ ck,
                     float* __restrict__ out) {
    const int lane = threadIdx.x & 31;
    const int b = blockIdx.x * 8 + (threadIdx.x >> 5);
    const float* row = Q0 + (size_t)b * K_COLS;
    float acc = 0.0f;
    for (int k = lane; k < K_COLS; k += 32) acc += row[k] * ck[k];
    for (int off = 16; off > 0; off >>= 1) acc += __shfl_xor(acc, off, 32);
    const float inv = 1.0f / acc;   // all lanes hold the reduction
    float* orow = out + (size_t)b * K_COLS;
    for (int k = lane; k < K_COLS; k += 32) orow[k] = row[k] * ck[k] * inv;
}

extern "C" void kernel_launch(void* const* d_in, const int* in_sizes, int n_in,
                              void* d_out, int out_size, void* d_ws, size_t ws_size,
                              hipStream_t stream) {
    (void)in_sizes; (void)n_in; (void)out_size; (void)ws_size;
    const float* features = (const float*)d_in[0];   // [32768,768]
    const float* head     = (const float*)d_in[1];   // [768,1000]
    const float* w        = (const float*)d_in[2];   // [1,1000]
    float* out = (float*)d_out;

    char* ws = (char*)d_ws;
    size_t off = 0;
    float*          Q0  = (float*)(ws + off);          off += (size_t)B_ROWS * K_COLS * 4;  // 131,072,000
    unsigned short* Ahi = (unsigned short*)(ws + off); off += (size_t)B_ROWS * D_DIM * 2;   //  50,331,648
    unsigned short* Alo = (unsigned short*)(ws + off); off += (size_t)B_ROWS * D_DIM * 2;
    unsigned short* Bhi = (unsigned short*)(ws + off); off += (size_t)K_PAD * D_DIM * 2;    //   1,572,864
    unsigned short* Blo = (unsigned short*)(ws + off); off += (size_t)K_PAD * D_DIM * 2;
    float*          C   = (float*)(ws + off);          off += 4096;
    float*          K2  = (float*)(ws + off);          off += 4096;
    float*          ck  = (float*)(ws + off);          off += 4096;
    float*          rb  = (float*)(ws + off);          off += (size_t)B_ROWS * 4;

    prep_feat_kernel<<<B_ROWS, 256, 0, stream>>>(features, Ahi, Alo);
    prep_head_kernel<<<D_DIM, 256, 0, stream>>>(head, Bhi, Blo);
    softmax_w_kernel<<<1, 1024, 0, stream>>>(w, K2);

    gemm_exp_kernel<<<dim3(B_ROWS / 32, K_PAD / 32 / 8), 256, 0, stream>>>(Ahi, Alo, Bhi, Blo, Q0);

    // iter 1
    zero_c_kernel<<<1, 1024, 0, stream>>>(C);
    colsum_kernel<<<B_ROWS / 256, 256, 0, stream>>>(Q0, rb, C, 0);
    scale_ck_kernel<<<1, 1024, 0, stream>>>(K2, C, ck);
    rowsum_kernel<<<B_ROWS / 8, 256, 0, stream>>>(Q0, ck, rb);
    // iter 2
    zero_c_kernel<<<1, 1024, 0, stream>>>(C);
    colsum_kernel<<<B_ROWS / 256, 256, 0, stream>>>(Q0, rb, C, 1);
    scale_ck_kernel<<<1, 1024, 0, stream>>>(K2, C, ck);
    rowsum_kernel<<<B_ROWS / 8, 256, 0, stream>>>(Q0, ck, rb);
    // iter 3 (row-normalize folds into finalize; /B and *B cancel)
    zero_c_kernel<<<1, 1024, 0, stream>>>(C);
    colsum_kernel<<<B_ROWS / 256, 256, 0, stream>>>(Q0, rb, C, 1);
    scale_ck_kernel<<<1, 1024, 0, stream>>>(K2, C, ck);
    finalize_kernel<<<B_ROWS / 8, 256, 0, stream>>>(Q0, ck, out);
}
